// Q_class_embed_12876311953739
// MI455X (gfx1250) — 8-bit matrix operand order repaired and checked by reading, compiled, not run on hardware
//
#include <hip/hip_runtime.h>
#include <hip/hip_bf16.h>

typedef __attribute__((ext_vector_type(8))) int v8i;
typedef __attribute__((ext_vector_type(4))) int v4i;

#define QEPS 1e-8f

__global__ void k_init(unsigned* __restrict__ absmax) {
    if (threadIdx.x == 0 && blockIdx.x == 0) *absmax = 0u;
}

__global__ void k_absmax(const float4* __restrict__ hs4, long n4,
                         unsigned* __restrict__ absmax) {
    __shared__ float smax[256];
    float m = 0.f;
    long stride = (long)gridDim.x * blockDim.x;
    for (long i = (long)blockIdx.x * blockDim.x + threadIdx.x; i < n4; i += stride) {
        float4 v = hs4[i];
        m = fmaxf(m, fmaxf(fmaxf(fabsf(v.x), fabsf(v.y)),
                           fmaxf(fabsf(v.z), fabsf(v.w))));
    }
    smax[threadIdx.x] = m;
    __syncthreads();
    for (int s = 128; s > 0; s >>= 1) {
        if ((int)threadIdx.x < s)
            smax[threadIdx.x] = fmaxf(smax[threadIdx.x], smax[threadIdx.x + s]);
        __syncthreads();
    }
    if (threadIdx.x == 0) atomicMax(absmax, __float_as_uint(smax[0]));
}

__device__ __forceinline__ int q8(float x, float inv) {
    float r = rintf(x * inv);
    r = fminf(fmaxf(r, -128.f), 127.f);
    return (int)r;
}

__device__ __forceinline__ long a_swz_dword(long row, int k, int KT) {
    int tileRow = (int)(row >> 4), rr = (int)(row & 15);
    int tileK = k >> 6, kk = k & 63;
    int chunk = kk >> 4;
    int khalf = chunk & 1;
    int vg    = chunk >> 1;
    int lane  = khalf * 16 + rr;
    return (((long)tileRow * KT + tileK) << 8) + lane * 8 + vg * 4 + ((kk >> 2) & 3);
}

__device__ __forceinline__ long b_swz_dword(int col, int k, int KT) {
    int tileCol = col >> 4, cc = col & 15;
    int tileK = k >> 6, kk = k & 63;
    int seg    = kk >> 4;
    int lhalf  = seg & 1;
    int bgroup = seg >> 1;
    int lane   = lhalf * 16 + cc;
    return (((long)tileCol * KT + tileK) << 8) + lane * 8 + bgroup * 4 + ((kk >> 2) & 3);
}

__global__ void k_wquant(const float* __restrict__ weight, const float* __restrict__ bias,
                         const unsigned* __restrict__ absmax_bits,
                         int* __restrict__ w_swz,
                         float* __restrict__ bias_scale, float* __restrict__ b_int,
                         int D) {
    __shared__ float smax[256];
    int o = blockIdx.x, t = threadIdx.x;
    const float4* wr = (const float4*)(weight + (long)o * D);
    float4 wv = wr[t];
    float m = fmaxf(fmaxf(fabsf(wv.x), fabsf(wv.y)), fmaxf(fabsf(wv.z), fabsf(wv.w)));
    smax[t] = m;
    __syncthreads();
    for (int s = 128; s > 0; s >>= 1) {
        if (t < s) smax[t] = fmaxf(smax[t], smax[t + s]);
        __syncthreads();
    }
    float w_scale   = fmaxf(smax[0], QEPS) * (1.f / 127.f);
    float act_scale = fmaxf(__uint_as_float(*absmax_bits), QEPS) * (1.f / 127.f);
    if (t == 0) {
        float bs = w_scale * act_scale;
        bias_scale[o] = bs;
        b_int[o]      = rintf(bias[o] / bs);
    }
    float inv = 1.f / w_scale;
    int q0 = q8(wv.x, inv), q1 = q8(wv.y, inv), q2 = q8(wv.z, inv), q3 = q8(wv.w, inv);
    int packed = (q0 & 255) | ((q1 & 255) << 8) | ((q2 & 255) << 16) | ((q3 & 255) << 24);
    w_swz[b_swz_dword(o, t << 2, D >> 6)] = packed;
}

__global__ void k_aquant(const float4* __restrict__ hs4, int* __restrict__ x_swz,
                         const unsigned* __restrict__ absmax_bits, long n4, int K) {
    float scale = fmaxf(__uint_as_float(*absmax_bits), QEPS) * (1.f / 127.f);
    float inv = 1.f / scale;
    int kdw = K >> 2;
    int KT  = K >> 6;
    long stride = (long)gridDim.x * blockDim.x;
    for (long i = (long)blockIdx.x * blockDim.x + threadIdx.x; i < n4; i += stride) {
        float4 v = hs4[i];
        int q0 = q8(v.x, inv), q1 = q8(v.y, inv), q2 = q8(v.z, inv), q3 = q8(v.w, inv);
        int packed = (q0 & 255) | ((q1 & 255) << 8) | ((q2 & 255) << 16) | ((q3 & 255) << 24);
        long row = i / kdw;
        int  k   = (int)(i - row * kdw) << 2;
        x_swz[a_swz_dword(row, k, KT)] = packed;
    }
}

__global__ __launch_bounds__(256, 1)
void k_gemm(const v4i* __restrict__ A,
            const v4i* __restrict__ B,
            const float* __restrict__ bias_scale,
            const float* __restrict__ b_int,
            float* __restrict__ out,
            int M, int N, int K) {
    int tid   = threadIdx.x;
    int wave  = tid >> 5;
    int lane  = tid & 31;
    int l16   = lane & 15;
    int lhalf = lane >> 4;
    int waveM = wave & 1;
    int waveN = wave >> 1;
    int KT  = K >> 6;
    int mT0 = blockIdx.x * 8  + waveM * 4;
    int nT0 = blockIdx.y * 16 + waveN * 4;

    v8i acc[4][4];
#pragma unroll
    for (int i = 0; i < 4; ++i)
#pragma unroll
        for (int j = 0; j < 4; ++j) acc[i][j] = (v8i)0;

    const v4i* aW = A + (long)mT0 * KT * 64 + lane * 2;
    const v4i* bW = B + (long)nT0 * KT * 64 + lane * 2;

    for (int kt = 0; kt < KT; ++kt) {
        v8i af[4], bf[4];
#pragma unroll
        for (int ms = 0; ms < 4; ++ms) {
            const v4i* p = aW + ((long)ms * KT + kt) * 64;
            union { v4i h[2]; v8i v; } u;
            u.h[0] = p[0];
            u.h[1] = p[1];
            af[ms] = u.v;
        }
#pragma unroll
        for (int ns = 0; ns < 4; ++ns) {
            const v4i* p = bW + ((long)ns * KT + kt) * 64;
            union { v4i h[2]; v8i v; } u;
            u.h[0] = p[0];
            u.h[1] = p[1];
            bf[ns] = u.v;
        }
#pragma unroll
        for (int ms = 0; ms < 4; ++ms)
#pragma unroll
            for (int ns = 0; ns < 4; ++ns)
                acc[ms][ns] = __builtin_amdgcn_wmma_i32_16x16x64_iu8(
                     true, af[ms],  true, bf[ns],
                    acc[ms][ns],  false,  false);
    }

#pragma unroll
    for (int ns = 0; ns < 4; ++ns) {
        int n = (nT0 + ns) * 16 + l16;
        float bsc = bias_scale[n];
        float bi  = b_int[n];
#pragma unroll
        for (int ms = 0; ms < 4; ++ms) {
            int mrow = (mT0 + ms) * 16 + lhalf * 8;
            float* orow = out + (long)mrow * N + n;
            v8i a = acc[ms][ns];
#pragma unroll
            for (int j = 0; j < 8; ++j) {
                float val = ((float)a[j] + bi) * bsc;
                __builtin_nontemporal_store(val, orow + (long)j * N);
            }
        }
    }
}

extern "C" void kernel_launch(void* const* d_in, const int* in_sizes, int n_in,
                              void* d_out, int out_size, void* d_ws, size_t ws_size,
                              hipStream_t stream) {
    (void)n_in; (void)out_size; (void)ws_size;
    const float* hs     = (const float*)d_in[0];
    const float* weight = (const float*)d_in[1];
    const float* bias   = (const float*)d_in[2];
    float* out = (float*)d_out;

    int  O = in_sizes[2];
    int  D = in_sizes[1] / O;
    long M = (long)in_sizes[0] / D;
    long nElem = (long)in_sizes[0];

    char* ws = (char*)d_ws;
    unsigned* absmax     = (unsigned*)ws;
    float*    bias_scale = (float*)(ws + 256);
    float*    b_int      = (float*)(ws + 256 + (size_t)O * 4);
    int*      w_swz      = (int*)(ws + 256 + (size_t)O * 8);
    size_t xq_off = 256 + (size_t)O * 8 + (size_t)O * D;
    xq_off = (xq_off + 255) & ~(size_t)255;
    int* x_swz = (int*)(ws + xq_off);

    k_init<<<1, 1, 0, stream>>>(absmax);
    k_absmax<<<1024, 256, 0, stream>>>((const float4*)hs, nElem / 4, absmax);
    k_wquant<<<O, 256, 0, stream>>>(weight, bias, absmax, w_swz, bias_scale, b_int, D);
    k_aquant<<<2048, 256, 0, stream>>>((const float4*)hs, x_swz, absmax, nElem / 4, D);

    dim3 grid((unsigned)(M / 128), (unsigned)(O / 256));
    k_gemm<<<grid, 256, 0, stream>>>((const v4i*)x_swz, (const v4i*)w_swz,
                                     bias_scale, b_int, out, (int)M, O, D);
}
